// SaveAttnProcessor_40252433498545
// MI455X (gfx1250) — matrix operand order repaired and checked by reading, not compiled, not run on hardware
//
#include <hip/hip_runtime.h>
#include <stdint.h>

// ---------------- problem constants ----------------
#define BN      8
#define CC      512
#define SS      1024            // H*W = 32*32
#define GROUPS  32
#define CPG     16              // C / GROUPS
#define HEADS   8
#define DH      64              // C / HEADS
#define EPSV    1e-5f

typedef __attribute__((ext_vector_type(16))) _Float16 v16h;
typedef __attribute__((ext_vector_type(8)))  float    v8f;

union Frag16 { v16h v; uint32_t u[8]; };
union FragC  { v8f  v; float    f[8]; };
union H8     { uint4 u4; _Float16 h[8]; };

// ---------------- WMMA fragment loaders (ISA 7.12.2 layouts) ----------------
// A (f16 16x32, MxK): lane<16 -> row M=lane, K in {0..7, 16..23};
//                     lane>=16 -> row M=lane-16, K in {8..15, 24..31}
__device__ __forceinline__ void loadA(Frag16& a, const _Float16* base, int stride,
                                      int lane, int koff) {
    const uint32_t* p = (const uint32_t*)base;
    int m  = lane & 15;
    int kb = (lane >> 4) << 3;
#pragma unroll
    for (int j = 0; j < 8; ++j) {
        int k0 = ((j >> 2) << 4) + kb + ((j & 3) << 1) + koff;
        a.u[j] = p[(m * stride + k0) >> 1];
    }
}
// B (f16 32x16, KxN): lane<16 -> col N=lane, K in {0..7, 16..23} ; lane>=16 -> col N=lane-16, K in {8..15, 24..31}
// LDS layout expected: [N rows][K contiguous]
__device__ __forceinline__ void loadB(Frag16& b, const _Float16* base, int stride,
                                      int lane, int noff, int koff) {
    const uint32_t* p = (const uint32_t*)base;
    int n  = (lane & 15) + noff;
    int kb = (lane >> 4) << 3;
#pragma unroll
    for (int j = 0; j < 8; ++j) {
        int k0 = ((j >> 2) << 4) + kb + ((j & 3) << 1) + koff;
        b.u[j] = p[(n * stride + k0) >> 1];
    }
}

// ---------------- 1) GroupNorm statistics ----------------
__global__ __launch_bounds__(256) void gn_stats(const float* __restrict__ x,
                                                float* __restrict__ muRs) {
    int b = blockIdx.x / GROUPS, g = blockIdx.x % GROUPS;
    const float* base = x + ((size_t)b * CC + (size_t)g * CPG) * SS;
    float s = 0.f, s2 = 0.f;
    for (int i = threadIdx.x; i < CPG * SS; i += 256) {
        float v = base[i];
        s += v; s2 += v * v;
    }
    __shared__ float r1[256], r2[256];
    r1[threadIdx.x] = s; r2[threadIdx.x] = s2;
    __syncthreads();
    for (int off = 128; off > 0; off >>= 1) {
        if (threadIdx.x < off) {
            r1[threadIdx.x] += r1[threadIdx.x + off];
            r2[threadIdx.x] += r2[threadIdx.x + off];
        }
        __syncthreads();
    }
    if (threadIdx.x == 0) {
        const float inv = 1.0f / (float)(CPG * SS);
        float mu  = r1[0] * inv;
        float var = r2[0] * inv - mu * mu;
        muRs[blockIdx.x * 2 + 0] = mu;
        muRs[blockIdx.x * 2 + 1] = rsqrtf(var + EPSV);
    }
}

// ---------------- 2) transpose [B,C,S]->[B,S,C], emit x(f16) and normed(f16) ----------------
__global__ __launch_bounds__(256) void pack_xn(const float* __restrict__ x,
                                               const float* __restrict__ muRs,
                                               const float* __restrict__ gw,
                                               const float* __restrict__ gb,
                                               _Float16* __restrict__ xT,
                                               _Float16* __restrict__ nT) {
    __shared__ float tile[32][33];
    int b  = blockIdx.z;
    int c0 = blockIdx.y * 32;
    int s0 = blockIdx.x * 32;
    int tx = threadIdx.x & 31, ty = threadIdx.x >> 5;
    for (int r = ty; r < 32; r += 8)
        tile[r][tx] = x[((size_t)b * CC + c0 + r) * SS + s0 + tx];
    __syncthreads();
    for (int r = ty; r < 32; r += 8) {       // r = local s, tx = local c
        int c = c0 + tx;
        float v = tile[tx][r];
        int g = c >> 4;
        float mu = muRs[(b * GROUPS + g) * 2 + 0];
        float rs = muRs[(b * GROUPS + g) * 2 + 1];
        size_t o = ((size_t)b * SS + (s0 + r)) * CC + c;
        xT[o] = (_Float16)v;
        nT[o] = (_Float16)((v - mu) * rs * gw[c] + gb[c]);
    }
}

// ---------------- 3) fp32 -> f16 weight conversion ----------------
__global__ void cvt_f16(const float* __restrict__ s, _Float16* __restrict__ d, int n) {
    for (int i = blockIdx.x * blockDim.x + threadIdx.x; i < n; i += gridDim.x * blockDim.x)
        d[i] = (_Float16)s[i];
}

// ---------------- 4) WMMA projection GEMM: O[b,s,n] = A[b,s,:] . W[n,:] * scale ----------------
// per-wave tile 16(M) x 64(N), K=512 in chunks of 32
// vmode==0: O is [B,S,C] f16 (scattered u16 epilogue)
// vmode==1: O is [B,HEADS,DH,S] f16 (per-head transposed; one packed 16B store per N-tile)
__global__ __launch_bounds__(128) void wmma_gemm(const _Float16* __restrict__ Aact,
                                                 const _Float16* __restrict__ W,
                                                 _Float16* __restrict__ O,
                                                 float scale, int vmode) {
    __shared__ __align__(16) _Float16 As[4][16 * 32];
    __shared__ __align__(16) _Float16 Bs[4][64 * 32];
    int wid = threadIdx.x >> 5, lane = threadIdx.x & 31;
    int w    = blockIdx.x * 4 + wid;
    int n64  = w & 7;
    int s16  = (w >> 3) & 63;
    int b    = w >> 9;
    int s0 = s16 * 16, n0 = n64 * 64;

    FragC acc[4];
#pragma unroll
    for (int t = 0; t < 4; ++t)
#pragma unroll
        for (int j = 0; j < 8; ++j) acc[t].f[j] = 0.f;

    for (int kc = 0; kc < CC; kc += 32) {
        {   // A tile: 16 rows x 32 halves
            int row = lane & 15, part = lane >> 4;
            const uint4* g = (const uint4*)(Aact + ((size_t)(b * SS + s0 + row)) * CC + kc + part * 16);
            uint4* sp = (uint4*)(&As[wid][row * 32 + part * 16]);
            sp[0] = g[0]; sp[1] = g[1];
        }
        // B tile: rows = output channels n0..n0+63, K contiguous
        for (int rr = lane; rr < 64; rr += 32) {
            const uint4* g = (const uint4*)(W + (size_t)(n0 + rr) * CC + kc);
            uint4* sp = (uint4*)(&Bs[wid][rr * 32]);
            sp[0] = g[0]; sp[1] = g[1]; sp[2] = g[2]; sp[3] = g[3];
            if (kc + 32 < CC) __builtin_prefetch(W + (size_t)(n0 + rr) * CC + kc + 32, 0, 1);
        }
        __syncthreads();
        Frag16 af; loadA(af, As[wid], 32, lane, 0);
#pragma unroll
        for (int nt = 0; nt < 4; ++nt) {
            Frag16 bf; loadB(bf, Bs[wid], 32, lane, nt * 16, 0);
            acc[nt].v = __builtin_amdgcn_wmma_f32_16x16x32_f16(
                false, af.v, false, bf.v, (short)0, acc[nt].v, false, false);
        }
        __syncthreads();
    }
    int mb = (lane >> 4) << 3, nn = lane & 15;
    if (vmode) {
        // transposed per-head output: each lane owns 8 consecutive t for fixed channel n
#pragma unroll
        for (int nt = 0; nt < 4; ++nt) {
            int n = n0 + nt * 16 + nn;           // global channel = h*64 + dd
            H8 o;
#pragma unroll
            for (int i = 0; i < 8; ++i) o.h[i] = (_Float16)(acc[nt].f[i] * scale);
            *(uint4*)(O + ((size_t)((b * HEADS + (n >> 6)) * DH + (n & 63))) * SS + s0 + mb) = o.u4;
        }
    } else {
#pragma unroll
        for (int nt = 0; nt < 4; ++nt)
#pragma unroll
            for (int i = 0; i < 8; ++i)
                O[((size_t)(b * SS + s0 + mb + i)) * CC + n0 + nt * 16 + nn] =
                    (_Float16)(acc[nt].f[i] * scale);
    }
}

// ---------------- 5) flash attention: one (b,h,16-row) tile per wave ----------------
// q,k: [B,S,C] f16 ; vT: [B,HEADS,DH,S] f16 ; ctx: [B,S,C] f16
__global__ __launch_bounds__(128) void flash_attn(const _Float16* __restrict__ q,
                                                  const _Float16* __restrict__ k,
                                                  const _Float16* __restrict__ vT,
                                                  _Float16* __restrict__ ctx) {
    __shared__ __align__(16) _Float16 Qs[4][16 * 64];
    __shared__ __align__(16) _Float16 Ks[4][32 * 64];
    __shared__ __align__(16) _Float16 Vs[4][64 * 32];   // [dd][t] (already transposed in HBM)
    __shared__ __align__(16) _Float16 Ps[4][16 * 32];
    int wid = threadIdx.x >> 5, lane = threadIdx.x & 31;
    int w   = blockIdx.x * 4 + wid;
    int s16 = w & 63;
    int h   = (w >> 6) & 7;
    int b   = w >> 9;
    int s0 = s16 * 16;

    {   // Q tile 16 x 64
        int row = lane & 15, part = lane >> 4;
        const uint4* g = (const uint4*)(q + ((size_t)(b * SS + s0 + row)) * CC + h * DH + part * 32);
        uint4* sp = (uint4*)(&Qs[wid][row * 64 + part * 32]);
        sp[0] = g[0]; sp[1] = g[1]; sp[2] = g[2]; sp[3] = g[3];
    }
    __syncthreads();
    Frag16 qa[2];
    loadA(qa[0], Qs[wid], 64, lane, 0);
    loadA(qa[1], Qs[wid], 64, lane, 32);

    FragC acc[4];
#pragma unroll
    for (int t = 0; t < 4; ++t)
#pragma unroll
        for (int j = 0; j < 8; ++j) acc[t].f[j] = 0.f;
    float rowM[8], rowL[8];
#pragma unroll
    for (int i = 0; i < 8; ++i) { rowM[i] = -1e30f; rowL[i] = 0.f; }

    const _Float16* vhead = vT + ((size_t)(b * HEADS + h)) * DH * SS;

    for (int t0 = 0; t0 < SS; t0 += 32) {
        {   // K tile: 32 rows x 64 halves
            const uint4* g = (const uint4*)(k + ((size_t)(b * SS + t0 + lane)) * CC + h * DH);
            uint4* sp = (uint4*)(&Ks[wid][lane * 64]);
#pragma unroll
            for (int j = 0; j < 8; ++j) sp[j] = g[j];
        }
        // V tile: 64 channel-rows x 32 t-halves, straight coalesced copies
        for (int rr = lane; rr < 64; rr += 32) {
            const uint4* g = (const uint4*)(vhead + (size_t)rr * SS + t0);
            uint4* sp = (uint4*)(&Vs[wid][rr * 32]);
            sp[0] = g[0]; sp[1] = g[1]; sp[2] = g[2]; sp[3] = g[3];
        }
        __syncthreads();

        FragC sc[2];
#pragma unroll
        for (int t = 0; t < 2; ++t)
#pragma unroll
            for (int j = 0; j < 8; ++j) sc[t].f[j] = 0.f;
#pragma unroll
        for (int nt = 0; nt < 2; ++nt)
#pragma unroll
            for (int c = 0; c < 2; ++c) {
                Frag16 bk; loadB(bk, Ks[wid], 64, lane, nt * 16, c * 32);
                sc[nt].v = __builtin_amdgcn_wmma_f32_16x16x32_f16(
                    false, qa[c].v, false, bk.v, (short)0, sc[nt].v, false, false);
            }

        // online softmax over the 16x32 tile (rows live in 16-lane half-groups)
        int mrow = (lane >> 4) << 3;
        int ncol = lane & 15;
#pragma unroll
        for (int i = 0; i < 8; ++i) {
            float v0 = sc[0].f[i], v1 = sc[1].f[i];
            float mx = fmaxf(v0, v1);
            mx = fmaxf(mx, __shfl_xor(mx, 1));
            mx = fmaxf(mx, __shfl_xor(mx, 2));
            mx = fmaxf(mx, __shfl_xor(mx, 4));
            mx = fmaxf(mx, __shfl_xor(mx, 8));
            float mnew  = fmaxf(rowM[i], mx);
            float alpha = __expf(rowM[i] - mnew);
            float p0 = __expf(v0 - mnew);
            float p1 = __expf(v1 - mnew);
            float ps = p0 + p1;
            ps += __shfl_xor(ps, 1);
            ps += __shfl_xor(ps, 2);
            ps += __shfl_xor(ps, 4);
            ps += __shfl_xor(ps, 8);
            rowL[i] = rowL[i] * alpha + ps;
            rowM[i] = mnew;
            acc[0].f[i] *= alpha; acc[1].f[i] *= alpha;
            acc[2].f[i] *= alpha; acc[3].f[i] *= alpha;
            Ps[wid][(i + mrow) * 32 + ncol]      = (_Float16)p0;
            Ps[wid][(i + mrow) * 32 + ncol + 16] = (_Float16)p1;
        }
        __syncthreads();

        Frag16 pa; loadA(pa, Ps[wid], 32, lane, 0);
#pragma unroll
        for (int nt = 0; nt < 4; ++nt) {
            Frag16 bv; loadB(bv, Vs[wid], 32, lane, nt * 16, 0);
            acc[nt].v = __builtin_amdgcn_wmma_f32_16x16x32_f16(
                false, pa.v, false, bv.v, (short)0, acc[nt].v, false, false);
        }
        __syncthreads();
    }

    int mb = (lane >> 4) << 3, nn = lane & 15;
#pragma unroll
    for (int nt = 0; nt < 4; ++nt)
#pragma unroll
        for (int i = 0; i < 8; ++i)
            ctx[((size_t)(b * SS + s0 + mb + i)) * CC + h * DH + nt * 16 + nn] =
                (_Float16)(acc[nt].f[i] / rowL[i]);
}

// ---------------- 6) output GEMM (+bias), fp32, written directly as [B,C,S] ----------------
__global__ __launch_bounds__(128) void wmma_gemm_out(const _Float16* __restrict__ Aact,
                                                     const _Float16* __restrict__ W,
                                                     const float* __restrict__ bo,
                                                     float* __restrict__ out) {
    __shared__ __align__(16) _Float16 As[4][16 * 32];
    __shared__ __align__(16) _Float16 Bs[4][64 * 32];
    int wid = threadIdx.x >> 5, lane = threadIdx.x & 31;
    int w   = blockIdx.x * 4 + wid;
    int n64 = w & 7;
    int s16 = (w >> 3) & 63;
    int b   = w >> 9;
    int s0 = s16 * 16, n0 = n64 * 64;

    FragC acc[4];
#pragma unroll
    for (int t = 0; t < 4; ++t)
#pragma unroll
        for (int j = 0; j < 8; ++j) acc[t].f[j] = 0.f;

    for (int kc = 0; kc < CC; kc += 32) {
        {
            int row = lane & 15, part = lane >> 4;
            const uint4* g = (const uint4*)(Aact + ((size_t)(b * SS + s0 + row)) * CC + kc + part * 16);
            uint4* sp = (uint4*)(&As[wid][row * 32 + part * 16]);
            sp[0] = g[0]; sp[1] = g[1];
        }
        for (int rr = lane; rr < 64; rr += 32) {
            const uint4* g = (const uint4*)(W + (size_t)(n0 + rr) * CC + kc);
            uint4* sp = (uint4*)(&Bs[wid][rr * 32]);
            sp[0] = g[0]; sp[1] = g[1]; sp[2] = g[2]; sp[3] = g[3];
        }
        __syncthreads();
        Frag16 af; loadA(af, As[wid], 32, lane, 0);
#pragma unroll
        for (int nt = 0; nt < 4; ++nt) {
            Frag16 bf; loadB(bf, Bs[wid], 32, lane, nt * 16, 0);
            acc[nt].v = __builtin_amdgcn_wmma_f32_16x16x32_f16(
                false, af.v, false, bf.v, (short)0, acc[nt].v, false, false);
        }
        __syncthreads();
    }
    int mb = (lane >> 4) << 3, nn = lane & 15;
#pragma unroll
    for (int nt = 0; nt < 4; ++nt) {
        int n = n0 + nt * 16 + nn;
        float bias = bo[n];
        float* dst = out + ((size_t)b * CC + n) * SS + s0 + mb;
        float4 lo = { acc[nt].f[0] + bias, acc[nt].f[1] + bias,
                      acc[nt].f[2] + bias, acc[nt].f[3] + bias };
        float4 hi = { acc[nt].f[4] + bias, acc[nt].f[5] + bias,
                      acc[nt].f[6] + bias, acc[nt].f[7] + bias };
        *(float4*)(dst)     = lo;
        *(float4*)(dst + 4) = hi;
    }
}

// ---------------- host launcher ----------------
static inline size_t alignUp(size_t x) { return (x + 255) & ~(size_t)255; }

extern "C" void kernel_launch(void* const* d_in, const int* in_sizes, int n_in,
                              void* d_out, int out_size, void* d_ws, size_t ws_size,
                              hipStream_t stream) {
    (void)in_sizes; (void)n_in; (void)out_size; (void)ws_size;
    const float* x   = (const float*)d_in[0];
    const float* gnw = (const float*)d_in[1];
    const float* gnb = (const float*)d_in[2];
    const float* Wq  = (const float*)d_in[3];
    const float* Wk  = (const float*)d_in[4];
    const float* Wv  = (const float*)d_in[5];
    const float* Wo  = (const float*)d_in[6];
    const float* bo  = (const float*)d_in[7];
    float* out = (float*)d_out;

    char* p = (char*)d_ws;
    float*     muRs = (float*)p;       p += alignUp(BN * GROUPS * 2 * sizeof(float));
    _Float16*  xT   = (_Float16*)p;    p += alignUp((size_t)BN * SS * CC * 2);
    _Float16*  nT   = (_Float16*)p;    p += alignUp((size_t)BN * SS * CC * 2);
    _Float16*  Wq16 = (_Float16*)p;    p += alignUp((size_t)CC * CC * 2);
    _Float16*  Wk16 = (_Float16*)p;    p += alignUp((size_t)CC * CC * 2);
    _Float16*  Wv16 = (_Float16*)p;    p += alignUp((size_t)CC * CC * 2);
    _Float16*  Wo16 = (_Float16*)p;    p += alignUp((size_t)CC * CC * 2);
    _Float16*  qb   = (_Float16*)p;    p += alignUp((size_t)BN * SS * CC * 2);
    _Float16*  kb   = (_Float16*)p;    p += alignUp((size_t)BN * SS * CC * 2);
    _Float16*  vbT  = (_Float16*)p;    p += alignUp((size_t)BN * SS * CC * 2);  // [B,H,DH,S]
    _Float16*  ctx  = (_Float16*)p;    p += alignUp((size_t)BN * SS * CC * 2);

    gn_stats<<<BN * GROUPS, 256, 0, stream>>>(x, muRs);
    pack_xn<<<dim3(SS / 32, CC / 32, BN), 256, 0, stream>>>(x, muRs, gnw, gnb, xT, nT);
    cvt_f16<<<512, 256, 0, stream>>>(Wq, Wq16, CC * CC);
    cvt_f16<<<512, 256, 0, stream>>>(Wk, Wk16, CC * CC);
    cvt_f16<<<512, 256, 0, stream>>>(Wv, Wv16, CC * CC);
    cvt_f16<<<512, 256, 0, stream>>>(Wo, Wo16, CC * CC);

    const int gemmBlocks = BN * (SS / 16) * (CC / 64) / 4;   // 1024
    wmma_gemm<<<gemmBlocks, 128, 0, stream>>>(nT, Wq16, qb, 0.125f, 0);  // q (pre-scaled d^-0.5)
    wmma_gemm<<<gemmBlocks, 128, 0, stream>>>(xT, Wk16, kb, 1.0f, 0);    // k
    wmma_gemm<<<gemmBlocks, 128, 0, stream>>>(xT, Wv16, vbT, 1.0f, 1);   // v, per-head transposed

    const int attnBlocks = BN * HEADS * (SS / 16) / 4;       // 1024
    flash_attn<<<attnBlocks, 128, 0, stream>>>(qb, kb, vbT, ctx);

    wmma_gemm_out<<<gemmBlocks, 128, 0, stream>>>(ctx, Wo16, bo, out);
}
